// PosBiasedMultiHeadedAttention_63582695850885
// MI455X (gfx1250) — compile-verified, matrix operand order checked by reading, not run on hardware
//
#include <hip/hip_runtime.h>
#include <hip/hip_bf16.h>
#include <math.h>

// PosBiasedMultiHeadedAttention for MI455X (gfx1250), bf16 WMMA path.
//
// Bias factorization: sigmoid(lerp(Xtab,dx)@Wb + lerp(Ytab,dy)@Wb + bb)
//                   = sigmoid(lerp(Xtab@Wb,dx) + lerp(Ytab@Wb,dy) + bb)
// so T=x_table@Wb and U=y_table@Wb (41x12) are precomputed and the
// O(L^2 * PD) bias tensor never exists.
//
// All WMMA fragments are fetched with 16-byte ds_load_b128 accesses:
//  - A / K^T fragments: two contiguous 8-elem runs per lane in row-major LDS.
//  - W / V (B operand) tiles are staged into fragment-major LDS [nf][lane][16].

typedef __bf16 bf16;
typedef __attribute__((ext_vector_type(16))) __bf16 v16bf;
typedef __attribute__((ext_vector_type(8)))  __bf16 v8bf;
typedef __attribute__((ext_vector_type(4)))  __bf16 v4bf;
typedef __attribute__((ext_vector_type(2)))  __bf16 v2bf;
typedef __attribute__((ext_vector_type(8)))  float  v8f;

#define NH 12
#define DMODEL 768
#define DKH 64
#define LSEQ 512
#define NBATCH 2
#define NSTEPS 20
#define NTAB 41   // 2*NSTEPS+1
#define PDDIM 192

__device__ __forceinline__ v16bf combine16(v8bf lo, v8bf hi) {
  return __builtin_shufflevector(lo, hi, 0, 1, 2, 3, 4, 5, 6, 7,
                                 8, 9, 10, 11, 12, 13, 14, 15);
}

// A-operand (16x32 bf16) or K^T B-operand from row-major LDS:
// lane holds row (lane&15); k runs {kh..kh+7} and {kh+16..kh+23}, kh = 8*(lane>>4).
// Both runs are 16B-contiguous -> two ds_load_b128.
__device__ __forceinline__ v16bf load_row_frag(const bf16* p, int stride, int lane) {
  int m  = lane & 15;
  int kh = (lane & 16) ? 8 : 0;
  const bf16* base = p + m * stride + kh;
  v8bf lo = *(const v8bf*)(base);
  v8bf hi = *(const v8bf*)(base + 16);
  return combine16(lo, hi);
}

// B-operand from fragment-major staged LDS: 32 bytes contiguous per lane.
__device__ __forceinline__ v16bf load_fragmaj(const bf16* p, int lane) {
  const bf16* base = p + lane * 16;
  v8bf lo = *(const v8bf*)(base);
  v8bf hi = *(const v8bf*)(base + 8);
  return combine16(lo, hi);
}

__device__ __forceinline__ v8f wmma_bf16(v16bf a, v16bf b, v8f c) {
  return __builtin_amdgcn_wmma_f32_16x16x32_bf16(false, a, false, b, (short)0, c,
                                                 false, false);
}

// Intra-wave LDS producer->consumer ordering (CDNA5 split dep counters).
__device__ __forceinline__ void wave_lds_fence() {
  asm volatile("s_wait_dscnt 0" ::: "memory");
}

// k index for fragment-major B staging: register j -> source row k.
__device__ __forceinline__ int bfrag_k(int lane_t, int j) {
  int kh = (lane_t & 16) ? 8 : 0;
  return (j < 8) ? (kh + j) : (kh + 16 + (j - 8));
}

__device__ __forceinline__ float interp_tab(const float* t, float d) {
  float xs = fminf(fmaxf(d * ((float)NSTEPS / 1000.0f), -(float)NSTEPS), (float)NSTEPS);
  float lo = floorf(xs);
  float r  = xs - lo;
  int li = (int)lo + NSTEPS;
  int hi = li + 1; if (hi > 2 * NSTEPS) hi = 2 * NSTEPS;
  return t[li] * (1.0f - r) + t[hi] * r;
}

// ---------------------------------------------------------------------------
// Kernel 0: T = x_table @ Wb, U = y_table @ Wb   (each [41 x 12])
// ---------------------------------------------------------------------------
__global__ void bias_tables_kernel(const float* __restrict__ xt,
                                   const float* __restrict__ yt,
                                   const float* __restrict__ Wb,
                                   float* __restrict__ T, float* __restrict__ U) {
  int idx = blockIdx.x * blockDim.x + threadIdx.x;
  if (idx >= 2 * NTAB * NH) return;
  int which = (idx >= NTAB * NH) ? 1 : 0;
  int rem = which ? (idx - NTAB * NH) : idx;
  int li = rem / NH, h = rem % NH;
  const float* tab = which ? yt : xt;
  float s = 0.0f;
  for (int p = 0; p < PDDIM; ++p) s += tab[li * PDDIM + p] * Wb[p * NH + h];
  (which ? U : T)[li * NH + h] = s;
}

// ---------------------------------------------------------------------------
// bf16-WMMA GEMM: Y[M,N] = X[M,K] @ W[K,N] + bias[N]
// block = 128 threads (4 waves), tile 64x64, K-step 32.
// scatter==1: bf16 out in [B,H,L,DK]; scatter==0: f32 row-major out.
// ---------------------------------------------------------------------------
__global__ __launch_bounds__(128)
void gemm_bf16_wmma(const float* __restrict__ Xf, const bf16* __restrict__ Xb,
                    const float* __restrict__ W, const float* __restrict__ bias,
                    bf16* __restrict__ outb, float* __restrict__ outf,
                    int M, int N, int K, int scatter) {
  __shared__ __align__(16) bf16 Xs[64][40];        // row-major A tile
  __shared__ __align__(16) bf16 WsB[4][32][16];    // fragment-major B tile
  int tid  = threadIdx.x;
  int wave = tid >> 5, lane = tid & 31;
  int m0 = blockIdx.x * 64;
  int n0 = blockIdx.y * 64;

  v8f acc[4];
#pragma unroll
  for (int f = 0; f < 4; ++f)
    acc[f] = (v8f){0.f, 0.f, 0.f, 0.f, 0.f, 0.f, 0.f, 0.f};

  for (int k0 = 0; k0 < K; k0 += 32) {
    // Stage X tile (64x32) -> bf16, vectorized.
    if (Xf) {
      for (int i = tid; i < 512; i += 128) {
        int r = i >> 3, c4 = (i & 7) * 4;
        float4 x = *(const float4*)(Xf + (size_t)(m0 + r) * K + k0 + c4);
        v4bf o = {(bf16)x.x, (bf16)x.y, (bf16)x.z, (bf16)x.w};
        *(v4bf*)&Xs[r][c4] = o;
      }
    } else {
      for (int i = tid; i < 512; i += 128) {
        int r = i >> 3, c4 = (i & 7) * 4;
        *(v4bf*)&Xs[r][c4] = *(const v4bf*)(Xb + (size_t)(m0 + r) * K + k0 + c4);
      }
    }
    // Stage W tile (32x64) into fragment-major layout (pairwise stores).
    for (int i = tid; i < 4 * 8 * 32; i += 128) {
      int t  = i & 31;            // dest lane
      int jp = (i >> 5) & 7;      // register pair
      int nf = i >> 8;            // 16-col group
      int j  = jp * 2;
      int k  = bfrag_k(t, j);
      int n  = n0 + nf * 16 + (t & 15);
      float w0 = W[(size_t)(k0 + k) * N + n];
      float w1 = W[(size_t)(k0 + k + 1) * N + n];
      v2bf o = {(bf16)w0, (bf16)w1};
      *(v2bf*)&WsB[nf][t][j] = o;
    }
    __syncthreads();
    v16bf a = load_row_frag(&Xs[wave * 16][0], 40, lane);
#pragma unroll
    for (int f = 0; f < 4; ++f) {
      v16bf b = load_fragmaj(&WsB[f][0][0], lane);
      acc[f] = wmma_bf16(a, b, acc[f]);
    }
    __syncthreads();
  }

  int nlo = lane & 15;
  int mh  = (lane >> 4) * 8;
#pragma unroll
  for (int f = 0; f < 4; ++f) {
    float bv = bias[n0 + f * 16 + nlo];
#pragma unroll
    for (int r = 0; r < 8; ++r) {
      int m = m0 + wave * 16 + mh + r;
      int n = n0 + f * 16 + nlo;
      float v = acc[f][r] + bv;
      if (scatter) {
        int bi = m >> 9, l = m & (LSEQ - 1);
        int h = n >> 6, dk = n & (DKH - 1);
        outb[(((size_t)bi * NH + h) * LSEQ + l) * DKH + dk] = (bf16)v;
      } else {
        outf[(size_t)m * N + n] = v;
      }
    }
  }
}

// ---------------------------------------------------------------------------
// Flash-attention with fused positional bias.
// grid.x = B*H, grid.y = LQ/64. block = 128 (4 waves x 16 q-rows).
// ---------------------------------------------------------------------------
__global__ __launch_bounds__(128)
void attn_wmma(const bf16* __restrict__ qb, const bf16* __restrict__ kb,
               const bf16* __restrict__ vb,
               const float* __restrict__ qx, const float* __restrict__ qy,
               const float* __restrict__ kx, const float* __restrict__ ky,
               const float* __restrict__ T, const float* __restrict__ U,
               const float* __restrict__ bbias, bf16* __restrict__ ao) {
  __shared__ __align__(16) bf16 qs[64][72];          // row-major Q tile
  __shared__ __align__(16) bf16 ks[32][72];          // row-major K chunk (K^T frags)
  __shared__ __align__(16) bf16 VsB[4][32][16];      // fragment-major V chunk
  __shared__ __align__(16) bf16 ps[4][16][40];       // per-wave P (A-operand)
  __shared__ __align__(16) float sS[4][16][36];      // per-wave scores
  __shared__ __align__(16) float alpha_s[4][16];
  __shared__ __align__(16) float ssum_s[4][16];
  __shared__ float qx_s[64], qy_s[64], kx_s[32], ky_s[32];
  __shared__ float Tx_s[NTAB], Ty_s[NTAB];

  int tid  = threadIdx.x;
  int wave = tid >> 5, lane = tid & 31;
  int bh = blockIdx.x;
  int b = bh / NH, h = bh % NH;
  int q0 = blockIdx.y * 64;
  size_t head_base = ((size_t)b * NH + h) * (size_t)LSEQ * DKH;

  for (int i = tid; i < 64 * 8; i += 128) {   // 8x 16B chunks per 64-elem row
    int r = i >> 3, c8 = (i & 7) * 8;
    *(v8bf*)&qs[r][c8] = *(const v8bf*)(qb + head_base + (size_t)(q0 + r) * DKH + c8);
  }
  if (tid < 64) {
    qx_s[tid] = qx[b * LSEQ + q0 + tid];
    qy_s[tid] = qy[b * LSEQ + q0 + tid];
  }
  if (tid < NTAB) {
    Tx_s[tid] = T[tid * NH + h];
    Ty_s[tid] = U[tid * NH + h];
  }
  __syncthreads();

  float bbh = bbias[h];
  v16bf qa0 = load_row_frag(&qs[wave * 16][0], 72, lane);
  v16bf qa1 = load_row_frag(&qs[wave * 16][32], 72, lane);

  v8f O[4];
#pragma unroll
  for (int f = 0; f < 4; ++f)
    O[f] = (v8f){0.f, 0.f, 0.f, 0.f, 0.f, 0.f, 0.f, 0.f};
  float m_run = -1e30f, s_run = 0.0f;
  int nlo = lane & 15, mh = (lane >> 4) * 8;

  for (int kc = 0; kc < LSEQ; kc += 32) {
    __syncthreads();
    for (int i = tid; i < 32 * 8; i += 128) {
      int r = i >> 3, c8 = (i & 7) * 8;
      *(v8bf*)&ks[r][c8] = *(const v8bf*)(kb + head_base + (size_t)(kc + r) * DKH + c8);
    }
    for (int i = tid; i < 4 * 8 * 32; i += 128) {  // V -> fragment-major
      int t  = i & 31;
      int jp = (i >> 5) & 7;
      int nf = i >> 8;
      int j  = jp * 2;
      int k  = bfrag_k(t, j);
      int d  = nf * 16 + (t & 15);
      bf16 p0 = vb[head_base + (size_t)(kc + k) * DKH + d];
      bf16 p1 = vb[head_base + (size_t)(kc + k + 1) * DKH + d];
      v2bf o = {p0, p1};
      *(v2bf*)&VsB[nf][t][j] = o;
    }
    if (tid < 32) {
      kx_s[tid] = kx[b * LSEQ + kc + tid];
      ky_s[tid] = ky[b * LSEQ + kc + tid];
    }
    __syncthreads();

    // S = Q K^T / sqrt(DK) + sigmoid(bias), two 16-key groups per 32-chunk
#pragma unroll
    for (int g = 0; g < 2; ++g) {
      v8f s = (v8f){0.f, 0.f, 0.f, 0.f, 0.f, 0.f, 0.f, 0.f};
      v16bf b0 = load_row_frag(&ks[g * 16][0], 72, lane);   // d = 0..31
      v16bf b1 = load_row_frag(&ks[g * 16][32], 72, lane);  // d = 32..63
      s = wmma_bf16(qa0, b0, s);
      s = wmma_bf16(qa1, b1, s);
      float kxv = kx_s[g * 16 + nlo], kyv = ky_s[g * 16 + nlo];
#pragma unroll
      for (int r = 0; r < 8; ++r) {
        int m = mh + r;
        float logit = interp_tab(Tx_s, qx_s[wave * 16 + m] - kxv)
                    + interp_tab(Ty_s, qy_s[wave * 16 + m] - kyv) + bbh;
        float biasv = 1.0f / (1.0f + __expf(-logit));
        sS[wave][m][g * 16 + nlo] = s[r] * 0.125f + biasv;
      }
    }
    wave_lds_fence();

    // Online softmax: lanes 0..15 each own one q-row of this wave.
    if (lane < 16) {
      int row = lane;
      float rv[32];
      const float4* rp = (const float4*)&sS[wave][row][0];
#pragma unroll
      for (int c = 0; c < 8; ++c) {
        float4 fv = rp[c];
        rv[4 * c] = fv.x; rv[4 * c + 1] = fv.y;
        rv[4 * c + 2] = fv.z; rv[4 * c + 3] = fv.w;
      }
      float cmax = -1e30f;
#pragma unroll
      for (int j = 0; j < 32; ++j) cmax = fmaxf(cmax, rv[j]);
      float mnew  = fmaxf(m_run, cmax);
      float alpha = __expf(m_run - mnew);
      float rs = 0.0f;
      v8bf pk[4];
#pragma unroll
      for (int j = 0; j < 32; ++j) {
        float p = __expf(rv[j] - mnew);
        rs += p;
        pk[j >> 3][j & 7] = (bf16)p;
      }
#pragma unroll
      for (int c = 0; c < 4; ++c) *(v8bf*)&ps[wave][row][c * 8] = pk[c];
      s_run = s_run * alpha + rs;
      m_run = mnew;
      alpha_s[wave][row] = alpha;
    }
    wave_lds_fence();

    // Rescale running O by alpha(row), then O += P @ V
    {
      const float4* ap = (const float4*)&alpha_s[wave][mh];
      float4 a0 = ap[0], a1 = ap[1];
      float av[8] = {a0.x, a0.y, a0.z, a0.w, a1.x, a1.y, a1.z, a1.w};
#pragma unroll
      for (int r = 0; r < 8; ++r) {
#pragma unroll
        for (int f = 0; f < 4; ++f) O[f][r] *= av[r];
      }
    }
    v16bf pa = load_row_frag(&ps[wave][0][0], 40, lane);
#pragma unroll
    for (int f = 0; f < 4; ++f) {
      v16bf vbf = load_fragmaj(&VsB[f][0][0], lane);
      O[f] = wmma_bf16(pa, vbf, O[f]);
    }
  }

  if (lane < 16) ssum_s[wave][lane] = s_run;
  wave_lds_fence();

  {
    const float4* sp = (const float4*)&ssum_s[wave][mh];
    float4 s0 = sp[0], s1 = sp[1];
    float sv[8] = {s0.x, s0.y, s0.z, s0.w, s1.x, s1.y, s1.z, s1.w};
#pragma unroll
    for (int r = 0; r < 8; ++r) {
      float inv = 1.0f / sv[r];
      int qg = q0 + wave * 16 + mh + r;
#pragma unroll
      for (int f = 0; f < 4; ++f) {
        ao[((size_t)b * LSEQ + qg) * DMODEL + h * DKH + f * 16 + nlo] =
            (bf16)(O[f][r] * inv);
      }
    }
  }
}

// ---------------------------------------------------------------------------
// Host-side launcher
// ---------------------------------------------------------------------------
extern "C" void kernel_launch(void* const* d_in, const int* in_sizes, int n_in,
                              void* d_out, int out_size, void* d_ws, size_t ws_size,
                              hipStream_t stream) {
  const float* query = (const float*)d_in[0];
  const float* key   = (const float*)d_in[1];
  const float* value = (const float*)d_in[2];
  const float* qx = (const float*)d_in[3];
  const float* qy = (const float*)d_in[4];
  const float* kx = (const float*)d_in[5];
  const float* ky = (const float*)d_in[6];
  const float* Wq = (const float*)d_in[7];   const float* bq = (const float*)d_in[8];
  const float* Wk = (const float*)d_in[9];   const float* bk = (const float*)d_in[10];
  const float* Wv = (const float*)d_in[11];  const float* bv = (const float*)d_in[12];
  const float* Wo = (const float*)d_in[13];  const float* bo = (const float*)d_in[14];
  const float* xt = (const float*)d_in[15];
  const float* yt = (const float*)d_in[16];
  const float* Wb = (const float*)d_in[17];
  const float* bb = (const float*)d_in[18];

  char* ws = (char*)d_ws;
  const size_t HEAD_BYTES = (size_t)NBATCH * NH * LSEQ * DKH * sizeof(bf16); // 1.5 MB
  const size_t AO_BYTES   = (size_t)NBATCH * LSEQ * DMODEL * sizeof(bf16);   // 1.5 MB
  bf16* qbuf  = (bf16*)(ws);
  bf16* kbuf  = (bf16*)(ws + HEAD_BYTES);
  bf16* vbuf  = (bf16*)(ws + 2 * HEAD_BYTES);
  bf16* aobuf = (bf16*)(ws + 3 * HEAD_BYTES);
  float* Tt   = (float*)(ws + 3 * HEAD_BYTES + AO_BYTES);
  float* Ut   = Tt + NTAB * NH;

  const int M = NBATCH * LSEQ;           // 1024
  dim3 gproj(M / 64, DMODEL / 64);       // (16, 12)

  bias_tables_kernel<<<dim3(4), dim3(256), 0, stream>>>(xt, yt, Wb, Tt, Ut);
  gemm_bf16_wmma<<<gproj, 128, 0, stream>>>(query, nullptr, Wq, bq, qbuf, nullptr,
                                            M, DMODEL, DMODEL, 1);
  gemm_bf16_wmma<<<gproj, 128, 0, stream>>>(key, nullptr, Wk, bk, kbuf, nullptr,
                                            M, DMODEL, DMODEL, 1);
  gemm_bf16_wmma<<<gproj, 128, 0, stream>>>(value, nullptr, Wv, bv, vbuf, nullptr,
                                            M, DMODEL, DMODEL, 1);
  attn_wmma<<<dim3(NBATCH * NH, LSEQ / 64), 128, 0, stream>>>(
      qbuf, kbuf, vbuf, qx, qy, kx, ky, Tt, Ut, bb, aobuf);
  gemm_bf16_wmma<<<gproj, 128, 0, stream>>>(nullptr, aobuf, Wo, bo, nullptr,
                                            (float*)d_out, M, DMODEL, DMODEL, 0);
}
